// NTM_23776938950657
// MI455X (gfx1250) — matrix operand order repaired and checked by reading, not compiled, not run on hardware
//
#include <hip/hip_runtime.h>
#include <hip/hip_bf16.h>

// NTM cell, B=2048, IN=64, CTRL=512, M=1024, V=64, OUT=64.
// bf16 WMMA GEMM pipeline (v_wmma_f32_16x16x32_bf16, f32 accumulate).
// Per-wave 32-row M-blocking: every B-fragment feeds 2 (or 4) WMMAs.

typedef __attribute__((ext_vector_type(16))) __bf16 v16bf;
typedef __attribute__((ext_vector_type(8)))  __bf16 v8bf;
typedef __attribute__((ext_vector_type(8)))  float  v8f;

__device__ __forceinline__ __bf16 f2bf(float f) {
  unsigned int u = __builtin_bit_cast(unsigned int, f);
  unsigned int r = u + 0x7FFFu + ((u >> 16) & 1u);   // round-to-nearest-even
  unsigned short h = (unsigned short)(r >> 16);
  return __builtin_bit_cast(__bf16, h);
}

// A-fragment: 16x32 (MxK) tile at (row0,k0), row-major bf16, leading dim lda.
// lane: m=lane&15, half=lane>>4; elems 0..7 = K [half*8,+8), 8..15 = K [16+half*8,+8).
__device__ __forceinline__ v16bf load_a_bf(const __bf16* __restrict__ src, int lda,
                                           int row0, int k0, int lane) {
  const int half = (lane >> 4) & 1;
  const int m    = lane & 15;
  const __bf16* p = src + (size_t)(row0 + m) * lda + k0 + half * 8;
  v8bf lo = *(const v8bf*)(p);
  v8bf hi = *(const v8bf*)(p + 16);
  return __builtin_shufflevector(lo, hi, 0, 1, 2, 3, 4, 5, 6, 7,
                                 8, 9, 10, 11, 12, 13, 14, 15);
}

// B-fragment: 32x16 (KxN), B(k,n) = W[n0+n][k0+k], W row-major (N,K) bf16.
// lane: n=lane&15, half=lane>>4; elems 0..7 = K [half*8,+8), 8..15 = K [16+half*8,+8).
__device__ __forceinline__ v16bf load_b_bf(const __bf16* __restrict__ W, int ldw,
                                           int n0, int k0, int lane) {
  const int half = (lane >> 4) & 1;
  const int n    = lane & 15;
  const __bf16* p = W + (size_t)(n0 + n) * ldw + k0 + half * 8;
  v8bf lo = *(const v8bf*)(p);
  v8bf hi = *(const v8bf*)(p + 16);
  return __builtin_shufflevector(lo, hi, 0, 1, 2, 3, 4, 5, 6, 7,
                                 8, 9, 10, 11, 12, 13, 14, 15);
}

__device__ __forceinline__ v8f wmma_bf16(v16bf a, v16bf b, v8f c) {
  return __builtin_amdgcn_wmma_f32_16x16x32_bf16(false, a, false, b,
                                                 (short)0, c, false, false);
}

__device__ __forceinline__ float sigm(float x) { return 1.0f / (1.0f + __expf(-x)); }

// ---------------- prep: f32 -> bf16 conversions ----------------
__global__ __launch_bounds__(256) void k_f2bf(const float* __restrict__ src,
                                              __bf16* __restrict__ dst, int n) {
  int idx = blockIdx.x * blockDim.x + threadIdx.x;
  if (idx < n) dst[idx] = f2bf(src[idx]);
}

__global__ __launch_bounds__(256) void k_transpose_mem(const float* __restrict__ mem,
                                                       __bf16* __restrict__ memT) {
  int idx = blockIdx.x * blockDim.x + threadIdx.x;          // 1024*64
  if (idx < 1024 * 64) {
    int m = idx >> 6, v = idx & 63;
    memT[(size_t)v * 1024 + m] = f2bf(mem[idx]);
  }
}

__global__ __launch_bounds__(256) void k_copy_x(const float* __restrict__ x,
                                                __bf16* __restrict__ inp) {
  int idx = blockIdx.x * blockDim.x + threadIdx.x;          // 2048*64
  if (idx < 2048 * 64) {
    int b = idx >> 6, j = idx & 63;
    inp[(size_t)b * 128 + j] = f2bf(x[idx]);
  }
}

// ------- read_prev = rwp @ memory : (2048x1024)x(1024x64), 32-row M-block -------
__global__ __launch_bounds__(128) void k_readprev(const __bf16* __restrict__ rwp,
                                                  const __bf16* __restrict__ memT,
                                                  __bf16* __restrict__ inp) {
  const int lane = threadIdx.x & 31;
  const int tile = blockIdx.x * 4 + (threadIdx.x >> 5);     // 64*4 = 256 tiles
  const int nt = tile & 3, mt = tile >> 2;
  const int m0 = mt * 32, n0 = nt * 16;
  v8f cl = {}, ch = {};
  for (int k0 = 0; k0 < 1024; k0 += 32) {
    v16bf b = load_b_bf(memT, 1024, n0, k0, lane);
    cl = wmma_bf16(load_a_bf(rwp, 1024, m0,      k0, lane), b, cl);
    ch = wmma_bf16(load_a_bf(rwp, 1024, m0 + 16, k0, lane), b, ch);
  }
  const int half = (lane >> 4) & 1, n = lane & 15;
  for (int r = 0; r < 8; ++r) {
    const int row = m0 + r + half * 8;
    inp[(size_t)row * 128 + 64 + n0 + n]        = f2bf(cl[r]);
    inp[(size_t)(row + 16) * 128 + 64 + n0 + n] = f2bf(ch[r]);
  }
}

// ---------- gates + LSTM -> h (into hr[:, :512], bf16), 32-row M-block ----------
__global__ __launch_bounds__(256) void k_gates_lstm(
    const __bf16* __restrict__ inp, const __bf16* __restrict__ h_prev,
    const float* __restrict__ c_prev,
    const __bf16* __restrict__ W_ih, const float* __restrict__ b_ih,
    const __bf16* __restrict__ W_hh, const float* __restrict__ b_hh,
    __bf16* __restrict__ hr) {
  const int lane = threadIdx.x & 31;
  const int tile = blockIdx.x * 8 + (threadIdx.x >> 5);     // 64*32 = 2048 tiles
  const int nt = tile & 31, mt = tile >> 5;
  const int m0 = mt * 32, n0 = nt * 16;
  v8f g0l = {}, g1l = {}, g2l = {}, g3l = {};
  v8f g0h = {}, g1h = {}, g2h = {}, g3h = {};
#pragma unroll
  for (int k0 = 0; k0 < 128; k0 += 32) {
    v16bf al = load_a_bf(inp, 128, m0,      k0, lane);
    v16bf ah = load_a_bf(inp, 128, m0 + 16, k0, lane);
    v16bf b0 = load_b_bf(W_ih, 128, n0,        k0, lane);
    v16bf b1 = load_b_bf(W_ih, 128, n0 + 512,  k0, lane);
    v16bf b2 = load_b_bf(W_ih, 128, n0 + 1024, k0, lane);
    v16bf b3 = load_b_bf(W_ih, 128, n0 + 1536, k0, lane);
    g0l = wmma_bf16(al, b0, g0l); g0h = wmma_bf16(ah, b0, g0h);
    g1l = wmma_bf16(al, b1, g1l); g1h = wmma_bf16(ah, b1, g1h);
    g2l = wmma_bf16(al, b2, g2l); g2h = wmma_bf16(ah, b2, g2h);
    g3l = wmma_bf16(al, b3, g3l); g3h = wmma_bf16(ah, b3, g3h);
  }
  for (int k0 = 0; k0 < 512; k0 += 32) {
    v16bf al = load_a_bf(h_prev, 512, m0,      k0, lane);
    v16bf ah = load_a_bf(h_prev, 512, m0 + 16, k0, lane);
    v16bf b0 = load_b_bf(W_hh, 512, n0,        k0, lane);
    v16bf b1 = load_b_bf(W_hh, 512, n0 + 512,  k0, lane);
    v16bf b2 = load_b_bf(W_hh, 512, n0 + 1024, k0, lane);
    v16bf b3 = load_b_bf(W_hh, 512, n0 + 1536, k0, lane);
    g0l = wmma_bf16(al, b0, g0l); g0h = wmma_bf16(ah, b0, g0h);
    g1l = wmma_bf16(al, b1, g1l); g1h = wmma_bf16(ah, b1, g1h);
    g2l = wmma_bf16(al, b2, g2l); g2h = wmma_bf16(ah, b2, g2h);
    g3l = wmma_bf16(al, b3, g3l); g3h = wmma_bf16(ah, b3, g3h);
  }
  const int half = (lane >> 4) & 1, n = lane & 15;
  const int col = n0 + n;
  const float bi = b_ih[col]        + b_hh[col];
  const float bf = b_ih[col + 512]  + b_hh[col + 512];
  const float bg = b_ih[col + 1024] + b_hh[col + 1024];
  const float bo = b_ih[col + 1536] + b_hh[col + 1536];
  for (int r = 0; r < 8; ++r) {
    {
      const int row = m0 + r + half * 8;
      float ig = sigm(g0l[r] + bi);
      float fg = sigm(g1l[r] + bf);
      float gg = tanhf(g2l[r] + bg);
      float og = sigm(g3l[r] + bo);
      float c  = fg * c_prev[(size_t)row * 512 + col] + ig * gg;
      hr[(size_t)row * 576 + col] = f2bf(og * tanhf(c));
    }
    {
      const int row = m0 + 16 + r + half * 8;
      float ig = sigm(g0h[r] + bi);
      float fg = sigm(g1h[r] + bf);
      float gg = tanhf(g2h[r] + bg);
      float og = sigm(g3h[r] + bo);
      float c  = fg * c_prev[(size_t)row * 512 + col] + ig * gg;
      hr[(size_t)row * 576 + col] = f2bf(og * tanhf(c));
    }
  }
}

// -------- read/write logits + erase/add heads, 32-row M-block --------
__global__ __launch_bounds__(256) void k_heads(
    const __bf16* __restrict__ hr,
    const __bf16* __restrict__ W_read,  const float* __restrict__ b_read,
    const __bf16* __restrict__ W_write, const float* __restrict__ b_write,
    const __bf16* __restrict__ W_erase, const float* __restrict__ b_erase,
    const __bf16* __restrict__ W_add,   const float* __restrict__ b_add,
    float* __restrict__ rlog, float* __restrict__ wlog,
    float* __restrict__ ers,  float* __restrict__ adv) {
  const int lane = threadIdx.x & 31;
  const int tile = blockIdx.x * 8 + (threadIdx.x >> 5);     // 64*136 = 8704 tiles
  const int nt = tile % 136, mt = tile / 136;
  const int m0 = mt * 32;
  const __bf16* W; int n0;
  if (nt < 64)       { W = W_read;  n0 = nt * 16; }
  else if (nt < 128) { W = W_write; n0 = (nt - 64) * 16; }
  else if (nt < 132) { W = W_erase; n0 = (nt - 128) * 16; }
  else               { W = W_add;   n0 = (nt - 132) * 16; }
  v8f cl = {}, ch = {};
  for (int k0 = 0; k0 < 512; k0 += 32) {
    v16bf b = load_b_bf(W, 512, n0, k0, lane);
    cl = wmma_bf16(load_a_bf(hr, 576, m0,      k0, lane), b, cl);
    ch = wmma_bf16(load_a_bf(hr, 576, m0 + 16, k0, lane), b, ch);
  }
  const int half = (lane >> 4) & 1, n = lane & 15;
  const int col = n0 + n;
  for (int r = 0; r < 8; ++r) {
    const int row0 = m0 + r + half * 8;
    const int row1 = row0 + 16;
    if (nt < 64) {
      rlog[(size_t)row0 * 1024 + col] = cl[r] + b_read[col];
      rlog[(size_t)row1 * 1024 + col] = ch[r] + b_read[col];
    } else if (nt < 128) {
      wlog[(size_t)row0 * 1024 + col] = cl[r] + b_write[col];
      wlog[(size_t)row1 * 1024 + col] = ch[r] + b_write[col];
    } else if (nt < 132) {
      ers[(size_t)row0 * 64 + col] = sigm(cl[r] + b_erase[col]);
      ers[(size_t)row1 * 64 + col] = sigm(ch[r] + b_erase[col]);
    } else {
      adv[(size_t)row0 * 64 + col] = tanhf(cl[r] + b_add[col]);
      adv[(size_t)row1 * 64 + col] = tanhf(ch[r] + b_add[col]);
    }
  }
}

// -------- row softmax x2, rw2 = read_w*write_w (bf16 out), s = rowsum(rw2) --------
__global__ __launch_bounds__(256) void k_softmax(
    const float* __restrict__ rlog, const float* __restrict__ wlog,
    __bf16* __restrict__ read_w, __bf16* __restrict__ rw2, float* __restrict__ srow) {
  __shared__ float red[256];
  const int b = blockIdx.x, t = threadIdx.x;
  float r[4], w[4];
  for (int i = 0; i < 4; ++i) {
    r[i] = rlog[(size_t)b * 1024 + t + i * 256];
    w[i] = wlog[(size_t)b * 1024 + t + i * 256];
  }
  float mr = fmaxf(fmaxf(r[0], r[1]), fmaxf(r[2], r[3]));
  red[t] = mr; __syncthreads();
  for (int s = 128; s > 0; s >>= 1) { if (t < s) red[t] = fmaxf(red[t], red[t + s]); __syncthreads(); }
  mr = red[0]; __syncthreads();
  float mw = fmaxf(fmaxf(w[0], w[1]), fmaxf(w[2], w[3]));
  red[t] = mw; __syncthreads();
  for (int s = 128; s > 0; s >>= 1) { if (t < s) red[t] = fmaxf(red[t], red[t + s]); __syncthreads(); }
  mw = red[0]; __syncthreads();
  float er[4], ew[4], sr = 0.f, sw = 0.f;
  for (int i = 0; i < 4; ++i) {
    er[i] = __expf(r[i] - mr); sr += er[i];
    ew[i] = __expf(w[i] - mw); sw += ew[i];
  }
  red[t] = sr; __syncthreads();
  for (int s = 128; s > 0; s >>= 1) { if (t < s) red[t] += red[t + s]; __syncthreads(); }
  sr = red[0]; __syncthreads();
  red[t] = sw; __syncthreads();
  for (int s = 128; s > 0; s >>= 1) { if (t < s) red[t] += red[t + s]; __syncthreads(); }
  sw = red[0]; __syncthreads();
  const float ir = 1.f / sr, iw = 1.f / sw;
  float ps = 0.f;
  for (int i = 0; i < 4; ++i) {
    float rv = er[i] * ir, wv = ew[i] * iw, p = rv * wv;
    read_w[(size_t)b * 1024 + t + i * 256] = f2bf(rv);
    rw2[(size_t)b * 1024 + t + i * 256]    = f2bf(p);
    ps += p;
  }
  red[t] = ps; __syncthreads();
  for (int s = 128; s > 0; s >>= 1) { if (t < s) red[t] += red[t + s]; __syncthreads(); }
  if (t == 0) srow[b] = red[0];
}

// -- read_new = t1 - erase*t2 + add*s (into hr[:, 512:576], bf16), 32-row M-block --
__global__ __launch_bounds__(128) void k_readnew(
    const __bf16* __restrict__ read_w, const __bf16* __restrict__ rw2,
    const __bf16* __restrict__ memT, const float* __restrict__ ers,
    const float* __restrict__ adv, const float* __restrict__ srow,
    __bf16* __restrict__ hr) {
  const int lane = threadIdx.x & 31;
  const int tile = blockIdx.x * 4 + (threadIdx.x >> 5);     // 64*4 = 256 tiles
  const int nt = tile & 3, mt = tile >> 2;
  const int m0 = mt * 32, n0 = nt * 16;
  v8f t1l = {}, t2l = {}, t1h = {}, t2h = {};
  for (int k0 = 0; k0 < 1024; k0 += 32) {
    v16bf b = load_b_bf(memT, 1024, n0, k0, lane);
    t1l = wmma_bf16(load_a_bf(read_w, 1024, m0,      k0, lane), b, t1l);
    t1h = wmma_bf16(load_a_bf(read_w, 1024, m0 + 16, k0, lane), b, t1h);
    t2l = wmma_bf16(load_a_bf(rw2,    1024, m0,      k0, lane), b, t2l);
    t2h = wmma_bf16(load_a_bf(rw2,    1024, m0 + 16, k0, lane), b, t2h);
  }
  const int half = (lane >> 4) & 1, n = lane & 15;
  const int col = n0 + n;
  for (int r = 0; r < 8; ++r) {
    const int row0 = m0 + r + half * 8;
    const int row1 = row0 + 16;
    float v0 = t1l[r] - ers[(size_t)row0 * 64 + col] * t2l[r]
             + adv[(size_t)row0 * 64 + col] * srow[row0];
    float v1 = t1h[r] - ers[(size_t)row1 * 64 + col] * t2h[r]
             + adv[(size_t)row1 * 64 + col] * srow[row1];
    hr[(size_t)row0 * 576 + 512 + col] = f2bf(v0);
    hr[(size_t)row1 * 576 + 512 + col] = f2bf(v1);
  }
}

// ------ out = [h|read_new] @ W_out^T + b_out (f32 out), 32-row M-block ------
__global__ __launch_bounds__(128) void k_out(
    const __bf16* __restrict__ hr, const __bf16* __restrict__ W_out,
    const float* __restrict__ b_out, float* __restrict__ out) {
  const int lane = threadIdx.x & 31;
  const int tile = blockIdx.x * 4 + (threadIdx.x >> 5);     // 64*4 = 256 tiles
  const int nt = tile & 3, mt = tile >> 2;
  const int m0 = mt * 32, n0 = nt * 16;
  v8f cl = {}, ch = {};
  for (int k0 = 0; k0 < 576; k0 += 32) {
    v16bf b = load_b_bf(W_out, 576, n0, k0, lane);
    cl = wmma_bf16(load_a_bf(hr, 576, m0,      k0, lane), b, cl);
    ch = wmma_bf16(load_a_bf(hr, 576, m0 + 16, k0, lane), b, ch);
  }
  const int half = (lane >> 4) & 1, n = lane & 15;
  const int col = n0 + n;
  for (int r = 0; r < 8; ++r) {
    const int row = m0 + r + half * 8;
    out[(size_t)row * 64 + col]        = cl[r] + b_out[col];
    out[(size_t)(row + 16) * 64 + col] = ch[r] + b_out[col];
  }
}

extern "C" void kernel_launch(void* const* d_in, const int* in_sizes, int n_in,
                              void* d_out, int out_size, void* d_ws, size_t ws_size,
                              hipStream_t stream) {
  const float* x       = (const float*)d_in[0];
  const float* h_prev  = (const float*)d_in[1];
  const float* c_prev  = (const float*)d_in[2];
  const float* rwp     = (const float*)d_in[3];
  const float* memory  = (const float*)d_in[4];
  const float* W_ih    = (const float*)d_in[5];
  const float* b_ih    = (const float*)d_in[6];
  const float* W_hh    = (const float*)d_in[7];
  const float* b_hh    = (const float*)d_in[8];
  const float* W_read  = (const float*)d_in[9];
  const float* b_read  = (const float*)d_in[10];
  const float* W_write = (const float*)d_in[11];
  const float* b_write = (const float*)d_in[12];
  const float* W_erase = (const float*)d_in[13];
  const float* b_erase = (const float*)d_in[14];
  const float* W_add   = (const float*)d_in[15];
  const float* b_add   = (const float*)d_in[16];
  const float* W_out   = (const float*)d_in[17];
  const float* b_out   = (const float*)d_in[18];

  // f32 scratch
  float* wsf = (float*)d_ws;
  float* rlog = wsf; wsf += 2048 * 1024;
  float* wlog = wsf; wsf += 2048 * 1024;
  float* ers  = wsf; wsf += 2048 * 64;
  float* adv  = wsf; wsf += 2048 * 64;
  float* srow = wsf; wsf += 2048;
  // bf16 scratch (starts 16B aligned: preceding block is a multiple of 64 floats)
  __bf16* wsb = (__bf16*)wsf;
  __bf16* memT_bf   = wsb; wsb += 64 * 1024;
  __bf16* inp_bf    = wsb; wsb += 2048 * 128;
  __bf16* hr_bf     = wsb; wsb += 2048 * 576;
  __bf16* rwp_bf    = wsb; wsb += 2048 * 1024;
  __bf16* hprev_bf  = wsb; wsb += 2048 * 512;
  __bf16* readw_bf  = wsb; wsb += 2048 * 1024;
  __bf16* rw2_bf    = wsb; wsb += 2048 * 1024;
  __bf16* Wih_bf    = wsb; wsb += 2048 * 128;
  __bf16* Whh_bf    = wsb; wsb += 2048 * 512;
  __bf16* Wread_bf  = wsb; wsb += 1024 * 512;
  __bf16* Wwrite_bf = wsb; wsb += 1024 * 512;
  __bf16* Werase_bf = wsb; wsb += 64 * 512;
  __bf16* Wadd_bf   = wsb; wsb += 64 * 512;
  __bf16* Wout_bf   = wsb; wsb += 64 * 576;

  auto cvt = [&](const float* s, __bf16* d, int n) {
    k_f2bf<<<(n + 255) / 256, 256, 0, stream>>>(s, d, n);
  };
  cvt(W_ih,    Wih_bf,    2048 * 128);
  cvt(W_hh,    Whh_bf,    2048 * 512);
  cvt(W_read,  Wread_bf,  1024 * 512);
  cvt(W_write, Wwrite_bf, 1024 * 512);
  cvt(W_erase, Werase_bf, 64 * 512);
  cvt(W_add,   Wadd_bf,   64 * 512);
  cvt(W_out,   Wout_bf,   64 * 576);
  cvt(h_prev,  hprev_bf,  2048 * 512);
  cvt(rwp,     rwp_bf,    2048 * 1024);
  k_transpose_mem<<<256, 256, 0, stream>>>(memory, memT_bf);
  k_copy_x<<<512, 256, 0, stream>>>(x, inp_bf);

  k_readprev<<<64, 128, 0, stream>>>(rwp_bf, memT_bf, inp_bf);
  k_gates_lstm<<<256, 256, 0, stream>>>(inp_bf, hprev_bf, c_prev,
                                        Wih_bf, b_ih, Whh_bf, b_hh, hr_bf);
  k_heads<<<1088, 256, 0, stream>>>(hr_bf, Wread_bf, b_read, Wwrite_bf, b_write,
                                    Werase_bf, b_erase, Wadd_bf, b_add,
                                    rlog, wlog, ers, adv);
  k_softmax<<<2048, 256, 0, stream>>>(rlog, wlog, readw_bf, rw2_bf, srow);
  k_readnew<<<64, 128, 0, stream>>>(readw_bf, rw2_bf, memT_bf, ers, adv, srow, hr_bf);
  k_out<<<64, 128, 0, stream>>>(hr_bf, Wout_bf, b_out, (float*)d_out);
}
